// QuantLinear_76003741270251
// MI455X (gfx1250) — matrix operand order repaired and checked by reading, not compiled, not run on hardware
//
#include <hip/hip_runtime.h>
#include <stdint.h>

typedef _Float16 half2v __attribute__((ext_vector_type(2)));
typedef _Float16 half8  __attribute__((ext_vector_type(8)));
typedef _Float16 v16h   __attribute__((ext_vector_type(16)));
typedef float    v8f    __attribute__((ext_vector_type(8)));
typedef uint32_t u32;
typedef uint32_t u32x4  __attribute__((ext_vector_type(4)));
typedef uint32_t u32x8  __attribute__((ext_vector_type(8)));

#define BM 128
#define BN 128
#define BK 64
#define LDA 72   // 64 + 8-half pad: 144B row stride (16B aligned, bank-skewed)
#define LDB 72

// Issue a TDM load of a 128-row x 64-col f16 tile (row stride K elements) into
// LDS at lds_byte_addr, with hardware padding of 4 dwords after every 32 dwords
// (i.e. 8 pad halves after each 64-half row -> LDS row stride = LDA = 72).
// 2D tensor: descriptor groups 2/3 are NULL -> 2-operand tensor_load_to_lds.
__device__ __forceinline__ void tdm_load_A_tile(u32 lds_byte_addr,
                                                const _Float16* gsrc, int K)
{
    const uint64_t ga = (uint64_t)(uintptr_t)gsrc;
    u32x4 d0;
    d0[0] = 1u;                                            // count=1, user-mode load
    d0[1] = lds_byte_addr;                                 // lds_addr
    d0[2] = (u32)ga;                                       // global_addr[31:0]
    d0[3] = ((u32)(ga >> 32) & 0x01FFFFFFu) | (2u << 30);  // addr[56:32] | type=2
    u32x8 d1;
    d1[0] = (1u << 16)    // data_size = 2 bytes
          | (1u << 20)    // pad_enable
          | (4u << 22)    // pad_interval code 4 = 32 dwords (one 64-half row)
          | (3u << 25);   // pad_amount  code 3 = 4 dwords  (8 halves)
    d1[1] = (64u  << 16);  // abar_addr=0        | tensor_dim0[15:0]  = 64
    d1[2] = (128u << 16);  // tensor_dim0 hi = 0 | tensor_dim1[15:0]  = 128
    d1[3] = (64u  << 16);  // tensor_dim1 hi = 0 | tile_dim0          = 64
    d1[4] = 128u;          // tile_dim1 = 128    | tile_dim2          = 0
    d1[5] = (u32)K;        // tensor_dim0_stride[31:0] = K
    d1[6] = 0u;            // stride hi, dim1_stride lo (unused for 2D)
    d1[7] = 0u;
    asm volatile("tensor_load_to_lds %0, %1" :: "s"(d0), "s"(d1) : "memory");
}

__global__ __launch_bounds__(256)
void gptq_wmma_kernel(const _Float16* __restrict__ x,
                      const int*      __restrict__ qweight,
                      const _Float16* __restrict__ scales,
                      const int*      __restrict__ qzeros,
                      const int*      __restrict__ g_idx,
                      const _Float16* __restrict__ bias,
                      float*          __restrict__ out,
                      int M, int N, int K)
{
    __shared__ __align__(16) _Float16 smA[2][BM * LDA];   // double-buffered (TDM)
    __shared__ __align__(16) _Float16 smB[BN * LDB];

    const int tid   = threadIdx.x;
    const int lane  = tid & 31;
    const int wave  = tid >> 5;     // 0..7
    const int waveM = wave >> 2;    // 0..1  -> 64 rows of C
    const int waveN = wave & 3;     // 0..3  -> 32 cols of C
    const int lhalf = lane >> 4;    // 0/1
    const int l16   = lane & 15;

    const int m0 = blockIdx.y * BM;
    const int n0 = blockIdx.x * BN;
    const u32 Npack = (u32)N >> 3;

    // B-dequant assignment: word w = tid + 256*i ; n = w&127 (constant per
    // thread), kr = w>>7 = krBase + 2*i  (qweight row within the K-stage)
    const int nB     = tid & 127;
    const int krBase = tid >> 7;    // 0 or 1

    // Strength-reduced qweight addressing: running base pointer, constant row
    // offsets (i*2*N words) inside a stage, += 8*N words per stage.
    const int* qptr = qweight + (size_t)krBase * N + (n0 + nB);
    const size_t qStageStep = (size_t)8 * (u32)N;          // words per K-stage

    const u32 smA_base   = (u32)(uintptr_t)&smA[0][0];    // LDS byte offset
    const u32 smA_stride = (u32)(BM * LDA * sizeof(_Float16));

    v8f acc[4][2];
    #pragma unroll
    for (int i = 0; i < 4; ++i)
        #pragma unroll
        for (int j = 0; j < 2; ++j)
            acc[i][j] = v8f{};

    // Prime the TDM pipeline: stage 0 A-tile into buffer 0.
    if (wave == 0)
        tdm_load_A_tile(smA_base, x + (size_t)m0 * K, K);

    int cur = 0;
    for (int k0 = 0; k0 < K; k0 += BK) {
        const bool has_next = (k0 + BK) < K;

        // ---- fetch + dequant B tile into regs (per-thread column nB) ----
        const int g      = g_idx[k0];                      // stage lies in one group
        const u32 gN     = (u32)g * (u32)N;                // 32-bit group offsets
        const u32 gP     = (u32)g * Npack;
        const _Float16 sc    = scales[gN + (u32)(n0 + nB)];
        const int      zword = qzeros[gP + ((u32)(n0 + nB) >> 3)];
        const int      zp1   = ((zword >> ((nB & 7) * 4)) & 15) + 1;

        half2v s2;  s2[0] = sc;  s2[1] = sc;               // packed scale
        const _Float16 coff = (_Float16)(float)(1024 + zp1);
        half2v c2;  c2[0] = coff; c2[1] = coff;            // packed (1024 + z + 1)

        half8 bReg[4];
        #pragma unroll
        for (int i = 0; i < 4; ++i) {
            // row kr = krBase + 2*i -> constant word offset 2*N*i from qptr
            u32 q32 = (u32)qptr[(size_t)(2 * i) * (u32)N];
            half8 wv;
            #pragma unroll
            for (int p = 0; p < 4; ++p) {                  // nibble pair (2p, 2p+1)
                u32 t = q32 >> (8 * p);
                // f16 magic: (q | 0x6400) == 1024 + q ; pack two lanes
                u32 w = ((t & 0xFu) | ((t << 12) & 0x000F0000u)) | 0x64006400u;
                half2v h = __builtin_bit_cast(half2v, w);
                half2v r = (h - c2) * s2;                  // exact sub, one rounding
                wv[2 * p]     = r[0];
                wv[2 * p + 1] = r[1];
            }
            bReg[i] = wv;
        }

        if (has_next)   // prefetch next-stage qweight rows
            __builtin_prefetch(qptr + qStageStep, 0, 3);

        __syncthreads();   // previous stage's LDS reads retired

        #pragma unroll
        for (int i = 0; i < 4; ++i) {
            int kr = krBase + 2 * i;
            *(half8*)(smB + nB * LDB + kr * 8) = bReg[i];  // N-major (Bt)
        }

        if (wave == 0) {
            if (has_next)   // stream stage i+1 A-tile while we compute stage i
                tdm_load_A_tile(smA_base + (u32)(cur ^ 1) * smA_stride,
                                x + (size_t)m0 * K + (k0 + BK), K);
            if (has_next) __builtin_amdgcn_s_wait_tensorcnt(1);  // stage i landed
            else          __builtin_amdgcn_s_wait_tensorcnt(0);
        }
        __syncthreads();   // A(i) + smB visible to all waves

        const _Float16* __restrict__ As = &smA[cur][0];

        // ---- compute: two 16x16x32 WMMA K-steps per stage ----
        #pragma unroll
        for (int s = 0; s < 2; ++s) {
            // A frags (ISA 16-bit A 16x32): lane(m=l16,h=lhalf),
            // VGPR0-3: K = h*8 + 2v ; VGPR4-7: +16
            v16h aF[4];
            #pragma unroll
            for (int mt = 0; mt < 4; ++mt) {
                int m = waveM * 64 + mt * 16 + l16;
                const _Float16* p = As + m * LDA + s * 32 + lhalf * 8;
                half8 lo = *(const half8*)(p);
                half8 hi = *(const half8*)(p + 16);
                v16h a;
                #pragma unroll
                for (int e = 0; e < 8; ++e) { a[e] = lo[e]; a[e + 8] = hi[e]; }
                aF[mt] = a;
            }
            // B frags: lane(n=l16,h=lhalf): K = h*8 + 0..7, +16 of column n
            v16h bF[2];
            #pragma unroll
            for (int nt = 0; nt < 2; ++nt) {
                int n = waveN * 32 + nt * 16 + l16;
                const _Float16* p = smB + n * LDB + s * 32 + lhalf * 8;
                half8 lo = *(const half8*)(p);
                half8 hi = *(const half8*)(p + 16);
                v16h b;
                #pragma unroll
                for (int e = 0; e < 8; ++e) { b[e] = lo[e]; b[e + 8] = hi[e]; }
                bF[nt] = b;
            }
            #pragma unroll
            for (int mt = 0; mt < 4; ++mt)
                #pragma unroll
                for (int nt = 0; nt < 2; ++nt)
                    acc[mt][nt] = __builtin_amdgcn_wmma_f32_16x16x32_f16(
                        false, aF[mt], false, bF[nt],
                        (short)0, acc[mt][nt], false, false);
        }
        cur ^= 1;
        qptr += qStageStep;
    }

    // ---- epilogue: bias add + f32 store; C layout VGPR r -> M = r + 8*lhalf ----
    #pragma unroll
    for (int nt = 0; nt < 2; ++nt) {
        int nc = n0 + waveN * 32 + nt * 16 + l16;
        float bv = (float)bias[nc];
        #pragma unroll
        for (int mt = 0; mt < 4; ++mt) {
            int mBase = m0 + waveM * 64 + mt * 16 + lhalf * 8;
            v8f c = acc[mt][nt];
            #pragma unroll
            for (int r = 0; r < 8; ++r)
                out[(size_t)(mBase + r) * N + nc] = c[r] + bv;
        }
    }
}

extern "C" void kernel_launch(void* const* d_in, const int* in_sizes, int n_in,
                              void* d_out, int out_size, void* d_ws, size_t ws_size,
                              hipStream_t stream) {
    const _Float16* x       = (const _Float16*)d_in[0];
    const int*      qweight = (const int*)d_in[1];
    const _Float16* scales  = (const _Float16*)d_in[2];
    const int*      qzeros  = (const int*)d_in[3];
    const int*      g_idx   = (const int*)d_in[4];
    const _Float16* bias    = (const _Float16*)d_in[5];
    float*          out     = (float*)d_out;

    const int K = in_sizes[4];          // g_idx has K entries
    const int N = in_sizes[5];          // bias has N entries
    const int M = in_sizes[0] / K;      // x is [M, K]

    dim3 grid(N / BN, M / BM);
    gptq_wmma_kernel<<<grid, dim3(256), 0, stream>>>(
        x, qweight, scales, qzeros, g_idx, bias, out, M, N, K);
}
